// Multihead_Cross_attention_8366596292902
// MI455X (gfx1250) — matrix operand order repaired and checked by reading, not compiled, not run on hardware
//
#include <hip/hip_runtime.h>

// ---------------- CDNA5 WMMA types / helpers ----------------
typedef _Float16 v16h __attribute__((ext_vector_type(16)));
typedef _Float16 v8h  __attribute__((ext_vector_type(8)));
typedef float    v8f  __attribute__((ext_vector_type(8)));

__device__ __forceinline__ v16h cat8(v8h lo, v8h hi) {
  return __builtin_shufflevector(lo, hi, 0,1,2,3,4,5,6,7,8,9,10,11,12,13,14,15);
}

__device__ __forceinline__ v8f wmma16(v16h a, v16h b, v8f c) {
  // v_wmma_f32_16x16x32_f16
  return __builtin_amdgcn_wmma_f32_16x16x32_f16(false, a, false, b, (short)0, c, false, false);
}

// A-fragment (16x32 f16) from fp32 row-major A.
// p = &A[row*lda + k0 + (lane>=16 ? 16 : 0)] ; halves come from p[0..7] and p[8..15].
__device__ __forceinline__ v16h load_a_f32(const float* __restrict__ p) {
  float4 f0 = *(const float4*)(p);
  float4 f1 = *(const float4*)(p + 4);
  float4 f2 = *(const float4*)(p + 8);
  float4 f3 = *(const float4*)(p + 12);
  v16h a;
  a[0]=(_Float16)f0.x; a[1]=(_Float16)f0.y; a[2]=(_Float16)f0.z; a[3]=(_Float16)f0.w;
  a[4]=(_Float16)f1.x; a[5]=(_Float16)f1.y; a[6]=(_Float16)f1.z; a[7]=(_Float16)f1.w;
  a[8]=(_Float16)f2.x; a[9]=(_Float16)f2.y; a[10]=(_Float16)f2.z; a[11]=(_Float16)f2.w;
  a[12]=(_Float16)f3.x; a[13]=(_Float16)f3.y; a[14]=(_Float16)f3.z; a[15]=(_Float16)f3.w;
  return a;
}

// A-fragment from f16 row-major A (16-byte aligned rows).
__device__ __forceinline__ v16h load_a_f16(const _Float16* __restrict__ p) {
  v8h lo = *(const v8h*)p;
  v8h hh = *(const v8h*)(p + 8);
  return cat8(lo, hh);
}

// B-fragment from pre-swizzled "fragment-major" layout: 16 contiguous halfs per lane.
__device__ __forceinline__ v16h load_b_frag(const _Float16* __restrict__ p) {
  v8h lo = *(const v8h*)p;
  v8h hi = *(const v8h*)(p + 8);
  return cat8(lo, hi);
}

#define NEG_INF (-3.4028234663852886e38f)

// ---------------- Kernel 0: weight swizzle (f32 W[K][N] -> f16 fragment-major) ----
// Coalesced version: stage a 32x128 W tile in LDS (float4 row loads), then emit the
// fragment-major layout with contiguous 32B stores per thread.
// Wf[((kc*(N/16))+nt)*512 + lane*16 + e] = W[kc*32 + (lane>>4)*16 + e][nt*16 + (lane&15)]
__global__ __launch_bounds__(256)
void swizzle_w_kernel(const float* __restrict__ W, _Float16* __restrict__ Wf, int N) {
  const int kc = blockIdx.x;      // 32 K-rows
  const int nb = blockIdx.y;      // 128 N-cols
  const int tid = threadIdx.x;

  __shared__ __align__(16) _Float16 T[32][136];   // 32 x 128 (+8 pad)

  // coalesced load: 32 rows x 128 cols of f32 (each row = 32 float4s, 8 threads/row)
  {
    const int row = tid >> 3;          // 0..31
    const int c8  = tid & 7;           // 0..7
    const float* src = W + (size_t)(kc * 32 + row) * N + nb * 128;
    #pragma unroll
    for (int j = 0; j < 4; ++j) {
      int c4 = j * 8 + c8;             // float4 index within row
      float4 f = *(const float4*)(src + c4 * 4);
      T[row][c4 * 4 + 0] = (_Float16)f.x;
      T[row][c4 * 4 + 1] = (_Float16)f.y;
      T[row][c4 * 4 + 2] = (_Float16)f.z;
      T[row][c4 * 4 + 3] = (_Float16)f.w;
    }
  }
  __syncthreads();

  // emit 8 fragments (16 n-cols each), contiguous 32B store per thread
  {
    const int f    = tid >> 5;         // fragment 0..7
    const int lane = tid & 31;
    const int krow = (lane >> 4) << 4; // 0 or 16
    const int ncol = f * 16 + (lane & 15);
    v8h lo, hi;
    #pragma unroll
    for (int e = 0; e < 8; ++e) {
      lo[e] = T[krow + e][ncol];
      hi[e] = T[krow + 8 + e][ncol];
    }
    _Float16* dst = Wf + (((size_t)kc * (N >> 4) + nb * 8 + f) << 9) + lane * 16;
    *(v8h*)dst = lo;
    *(v8h*)(dst + 8) = hi;
  }
}

// ---------------- Kernel 1: generic WMMA GEMM ----------------
// C(MxN) = A(f32, row-major, lda=K) * B(f16 fragment-major) + bias
// 8 waves/WG, WG tile 256x64, wave tile 32x64 (2 row-groups x 4 n-tiles).
// mode 0: write f16 Q  -> Qh[b][h][q][64]
// mode 1: write f16 K/V -> Kf (scores-B layout) / Vf (PV-B layout)
// mode 2: write f32 out -> d_out, + padded-row zeroing
__global__ __launch_bounds__(256)
void wmma_gemm_kernel(const float* __restrict__ A, const _Float16* __restrict__ Bf,
                      const float* __restrict__ bias, int N, int K, int mode,
                      void* __restrict__ out0, void* __restrict__ out1,
                      const int* __restrict__ flags) {
  const int lane = threadIdx.x & 31;
  const int w    = threadIdx.x >> 5;
  const int l15  = lane & 15;
  const int hi   = lane >> 4;
  const int m0   = blockIdx.x * 256 + w * 32;
  const int nt0  = blockIdx.y * 4;        // base 16-wide n-tile
  const int ntiles = N >> 4;

  const float* arow0 = A + (size_t)(m0 + l15) * K + hi * 16;
  const float* arow1 = arow0 + (size_t)16 * K;
  v8f acc[2][4] = {};

  const int kcN = K >> 5;
  for (int kc = 0; kc < kcN; ++kc) {
    v16h a0 = load_a_f32(arow0 + kc * 32);
    v16h a1 = load_a_f32(arow1 + kc * 32);
    const _Float16* bp = Bf + (((size_t)kc * ntiles + nt0) << 9) + lane * 16;
    #pragma unroll
    for (int t = 0; t < 4; ++t) {
      v16h bfr = load_b_frag(bp + (size_t)t * 512);
      acc[0][t] = wmma16(a0, bfr, acc[0][t]);
      acc[1][t] = wmma16(a1, bfr, acc[1][t]);
    }
  }

  #pragma unroll
  for (int rg = 0; rg < 2; ++rg) {
    #pragma unroll
    for (int t = 0; t < 4; ++t) {
      int n = (nt0 + t) * 16 + l15;
      float bv = bias[n];
      #pragma unroll
      for (int r = 0; r < 8; ++r) {
        int m = m0 + rg * 16 + hi * 8 + r;
        float v = acc[rg][t][r] + bv;
        if (mode == 0) {
          int b = m >> 10, q = m & 1023, h = n >> 6, d = n & 63;
          _Float16* Qh = (_Float16*)out0;
          Qh[(((size_t)(b * 16 + h) * 1024 + q) << 6) + d] = (_Float16)v;
        } else if (mode == 1) {
          int b = m >> 10, key = m & 1023, h = n >> 7, c = n & 127;
          if (c < 64) {                      // K part -> scores-B fragment layout
            int dc = c >> 5, dd = c & 31;
            _Float16* Kf = (_Float16*)out0;
            size_t off = ((((size_t)(b * 16 + h) * 2 + dc) * 64 + (key >> 4)) << 9)
                       + ((((dd >> 4) << 4) + (key & 15)) << 4) + (dd & 15);
            Kf[off] = (_Float16)v;
          } else {                           // V part -> PV-B fragment layout
            int dv = c - 64;
            _Float16* Vf = (_Float16*)out1;
            size_t off = ((((size_t)(b * 16 + h) * 32 + (key >> 5)) * 4 + (dv >> 4)) << 9)
                       + (((((key >> 4) & 1) << 4) + (dv & 15)) << 4) + (key & 15);
            Vf[off] = (_Float16)v;
          }
        } else {
          float* O = (float*)out0;
          O[(size_t)m * N + n] = flags[m] ? 0.0f : v;
        }
      }
    }
  }
}

// ---------------- Kernel 2: fused attention (scores + mask + softmax + PV) -----
// Grid: (QN/16, H, B). 8 waves; wave w owns keys [w*128, w*128+128).
__global__ __launch_bounds__(256)
void attn_kernel(const _Float16* __restrict__ Qh, const _Float16* __restrict__ Kf,
                 const _Float16* __restrict__ Vf, const unsigned char* __restrict__ mask,
                 float* __restrict__ att, float* __restrict__ attn_out,
                 int* __restrict__ flags) {
  const int qt = blockIdx.x, h = blockIdx.y, b = blockIdx.z;
  const int lane = threadIdx.x & 31;
  const int w    = threadIdx.x >> 5;
  const int l15  = lane & 15;
  const int hiL  = lane >> 4;
  const int q0   = qt * 16;
  const int bh   = b * 16 + h;

  __shared__ __align__(16) _Float16 Plds[16][1032];   // 16 q rows x 1024 keys (+pad)
  __shared__ float redmax[16][8];
  __shared__ float redsum[16][8];
  __shared__ float Ored[4][32][8];

  // --- Q fragments (reused across all 8 key tiles of this wave) ---
  const _Float16* qbase = Qh + (((size_t)bh * 1024 + q0 + l15) << 6) + hiL * 16;
  v16h aq0 = load_a_f16(qbase);
  v16h aq1 = load_a_f16(qbase + 32);

  // --- scores: 8 tiles of 16x16 per wave (keys w*128 .. +127) ---
  v8f acc[8];
  #pragma unroll
  for (int nt = 0; nt < 8; ++nt) {
    int kt = w * 8 + nt;
    v8f c = {};
    const _Float16* kp0 = Kf + ((((size_t)bh * 2 + 0) * 64 + kt) << 9) + lane * 16;
    const _Float16* kp1 = Kf + ((((size_t)bh * 2 + 1) * 64 + kt) << 9) + lane * 16;
    c = wmma16(aq0, load_b_frag(kp0), c);
    c = wmma16(aq1, load_b_frag(kp1), c);
    acc[nt] = c;
  }

  // --- scale + mask ---
  const float scale = 0.125f;  // 1/sqrt(64)
  #pragma unroll
  for (int nt = 0; nt < 8; ++nt) {
    #pragma unroll
    for (int r = 0; r < 8; ++r) {
      int k = w * 128 + nt * 16 + l15;
      int q = q0 + hiL * 8 + r;
      float s = acc[nt][r] * scale;
      unsigned char mm = mask[((size_t)(b * 1024 + q)) * 1024 + k];
      acc[nt][r] = mm ? s : NEG_INF;
    }
  }

  // --- row max (local -> half-wave shfl -> cross-wave via LDS) ---
  float mloc[8];
  #pragma unroll
  for (int r = 0; r < 8; ++r) {
    float m = acc[0][r];
    #pragma unroll
    for (int nt = 1; nt < 8; ++nt) m = fmaxf(m, acc[nt][r]);
    m = fmaxf(m, __shfl_xor(m, 1));
    m = fmaxf(m, __shfl_xor(m, 2));
    m = fmaxf(m, __shfl_xor(m, 4));
    m = fmaxf(m, __shfl_xor(m, 8));
    mloc[r] = m;
  }
  if (l15 == 0) {
    #pragma unroll
    for (int r = 0; r < 8; ++r) redmax[hiL * 8 + r][w] = mloc[r];
  }
  __syncthreads();
  float rmax[8];
  #pragma unroll
  for (int r = 0; r < 8; ++r) {
    int row = hiL * 8 + r;
    float m = redmax[row][0];
    #pragma unroll
    for (int j = 1; j < 8; ++j) m = fmaxf(m, redmax[row][j]);
    rmax[r] = m;
  }

  // --- exp + row sum ---
  float sloc[8];
  #pragma unroll
  for (int r = 0; r < 8; ++r) {
    float s = 0.0f;
    #pragma unroll
    for (int nt = 0; nt < 8; ++nt) {
      float p = __expf(acc[nt][r] - rmax[r]);
      acc[nt][r] = p;
      s += p;
    }
    s += __shfl_xor(s, 1);
    s += __shfl_xor(s, 2);
    s += __shfl_xor(s, 4);
    s += __shfl_xor(s, 8);
    sloc[r] = s;
  }
  if (l15 == 0) {
    #pragma unroll
    for (int r = 0; r < 8; ++r) redsum[hiL * 8 + r][w] = sloc[r];
  }
  __syncthreads();
  float rinv[8];
  #pragma unroll
  for (int r = 0; r < 8; ++r) {
    int row = hiL * 8 + r;
    float s = 0.0f;
    #pragma unroll
    for (int j = 0; j < 8; ++j) s += redsum[row][j];
    rinv[r] = 1.0f / s;
  }

  // --- normalize, write att_prob (fp32, coalesced across half-wave), stage f16 in LDS ---
  float* arow_base = att + ((size_t)bh * 1024 + q0) * 1024;
  #pragma unroll
  for (int nt = 0; nt < 8; ++nt) {
    #pragma unroll
    for (int r = 0; r < 8; ++r) {
      int k = w * 128 + nt * 16 + l15;
      int row = hiL * 8 + r;
      float p = acc[nt][r] * rinv[r];
      arow_base[(size_t)row * 1024 + k] = p;
      Plds[row][k] = (_Float16)p;
    }
  }

  // --- padded-row flags (same for all h; computed once) ---
  if (h == 0 && w == 0 && l15 == 0) {
    #pragma unroll
    for (int r = 0; r < 8; ++r)
      flags[b * 1024 + q0 + hiL * 8 + r] = (rmax[r] < -1.0e37f) ? 1 : 0;
  }
  __syncthreads();

  // --- PV: O(16x64) = P(16x1024) @ V(1024x64); wave -> (dt = w&3, khalf = w>>2) ---
  const int dt = w & 3;
  const int kh = w >> 2;
  v8f o = {};
  for (int kc = kh * 16; kc < kh * 16 + 16; ++kc) {
    const _Float16* pp = &Plds[l15][kc * 32 + hiL * 16];
    v16h a = load_a_f16(pp);
    const _Float16* vp = Vf + ((((size_t)bh * 32 + kc) * 4 + dt) << 9) + lane * 16;
    o = wmma16(a, load_b_frag(vp), o);
  }
  if (kh == 1) {
    #pragma unroll
    for (int r = 0; r < 8; ++r) Ored[dt][lane][r] = o[r];
  }
  __syncthreads();
  if (kh == 0) {
    #pragma unroll
    for (int r = 0; r < 8; ++r) {
      float v = o[r] + Ored[dt][lane][r];
      int q = q0 + hiL * 8 + r;
      attn_out[(size_t)(b * 1024 + q) * 1024 + h * 64 + dt * 16 + l15] = v;
    }
  }
}

// ---------------- host: kernel_launch ----------------
extern "C" void kernel_launch(void* const* d_in, const int* in_sizes, int n_in,
                              void* d_out, int out_size, void* d_ws, size_t ws_size,
                              hipStream_t stream) {
  const int B = 8, QN = 1024, KN = 1024, H = 16, DQ = 64, DV = 64;
  const int QIN = 1024, KVIN = 1024, OUTN = 1024;

  const float*         q_in  = (const float*)d_in[0];
  const float*         kv_in = (const float*)d_in[1];
  const unsigned char* mask  = (const unsigned char*)d_in[2];   // jax bool -> 1 byte
  const float*         Wq    = (const float*)d_in[3];
  const float*         bq    = (const float*)d_in[4];
  const float*         Wkv   = (const float*)d_in[5];
  const float*         bkv   = (const float*)d_in[6];
  const float*         Wo    = (const float*)d_in[7];
  const float*         bo    = (const float*)d_in[8];

  float* out = (float*)d_out;                              // (B,QN,OUT)
  float* att = out + (size_t)B * QN * OUTN;                // (B,H,QN,KN)

  // workspace carve-up (256B aligned)
  char* ws = (char*)d_ws;
  size_t off = 0;
  auto alloc = [&](size_t bytes) { size_t r = off; off = (off + bytes + 255) & ~(size_t)255; return r; };
  _Float16* Wq_f  = (_Float16*)(ws + alloc((size_t)QIN * H * DQ * 2));            // 2MB
  _Float16* Wkv_f = (_Float16*)(ws + alloc((size_t)KVIN * H * (DQ + DV) * 2));    // 4MB
  _Float16* Wo_f  = (_Float16*)(ws + alloc((size_t)OUTN * OUTN * 2));             // 2MB
  _Float16* Qh    = (_Float16*)(ws + alloc((size_t)B * H * QN * DQ * 2));         // 16MB
  _Float16* Kf    = (_Float16*)(ws + alloc((size_t)B * H * DQ * KN * 2));         // 16MB
  _Float16* Vf    = (_Float16*)(ws + alloc((size_t)B * H * KN * DV * 2));         // 16MB
  float*    AO    = (float*)   (ws + alloc((size_t)B * QN * OUTN * 4));           // 32MB
  int*      flags = (int*)     (ws + alloc((size_t)B * QN * 4));

  // 1) swizzle weights into B-fragment-major f16 (LDS-tiled, coalesced)
  swizzle_w_kernel<<<dim3(QIN / 32, (H * DQ) / 128),        256, 0, stream>>>(Wq,  Wq_f,  H * DQ);
  swizzle_w_kernel<<<dim3(KVIN / 32, (H * (DQ + DV)) / 128), 256, 0, stream>>>(Wkv, Wkv_f, H * (DQ + DV));
  swizzle_w_kernel<<<dim3(OUTN / 32, OUTN / 128),           256, 0, stream>>>(Wo,  Wo_f,  OUTN);

  // 2) Q projection: (B*QN x 1024) @ (1024 x 1024) -> Qh[b][h][q][64]
  wmma_gemm_kernel<<<dim3((B * QN) / 256, (H * DQ) / 64), 256, 0, stream>>>(
      q_in, Wq_f, bq, H * DQ, QIN, 0, (void*)Qh, nullptr, nullptr);

  // 3) KV projection: (B*KN x 1024) @ (1024 x 2048) -> Kf (scores-B) + Vf (PV-B)
  wmma_gemm_kernel<<<dim3((B * KN) / 256, (H * (DQ + DV)) / 64), 256, 0, stream>>>(
      kv_in, Wkv_f, bkv, H * (DQ + DV), KVIN, 1, (void*)Kf, (void*)Vf, nullptr);

  // 4) fused attention: scores + mask + softmax (-> att_prob) + PV (-> AO), flags
  attn_kernel<<<dim3(QN / 16, H, B), 256, 0, stream>>>(
      Qh, Kf, Vf, mask, att, AO, flags);

  // 5) output projection: (B*QN x 1024) @ (1024 x 1024) + bo, zero padded rows
  wmma_gemm_kernel<<<dim3((B * QN) / 256, OUTN / 64), 256, 0, stream>>>(
      AO, Wo_f, bo, OUTN, OUTN, 2, (void*)out, nullptr, flags);
}
